// NonLocalAttention_45758581571816
// MI455X (gfx1250) — matrix operand order repaired and checked by reading, not compiled, not run on hardware
//
#include <hip/hip_runtime.h>
#include <hip/hip_bf16.h>
#include <stdint.h>

typedef __bf16 bf16_t;
typedef __attribute__((ext_vector_type(16))) __bf16 v16bf;
typedef __attribute__((ext_vector_type(8)))  __bf16 v8bf;
typedef __attribute__((ext_vector_type(8)))  float  v8f;

#define HW   4096
#define CQK  32
#define CV   64
#define BK   64      // keys per iteration
#define NWAVE 8
#define NBLK (HW / BK)

static __device__ __forceinline__ v8f vzero8() {
  v8f z;
#pragma unroll
  for (int i = 0; i < 8; ++i) z[i] = 0.0f;
  return z;
}

static __device__ __forceinline__ v16bf frag_from(const bf16_t* lo, const bf16_t* hi) {
  v8bf a = *(const v8bf*)lo;
  v8bf b = *(const v8bf*)hi;
  v16bf r;
#pragma unroll
  for (int i = 0; i < 8; ++i) { r[i] = a[i]; r[i + 8] = b[i]; }
  return r;
}

// A fragment: 16(M) x 32(K) bf16 from row-major source, element stride ld.
// lanes 0-15: row r, K[0..7]+K[16..23]; lanes 16-31: row r, K[8..15]+K[24..31]
static __device__ __forceinline__ v16bf load_a_frag(const bf16_t* base, int ld, int r, int h) {
  const bf16_t* p = base + (size_t)r * ld + h * 8;
  return frag_from(p, p + 16);
}

// B fragment: 32(K) x 16(N) bf16; source holds B^T row-major (row = column n).
// lanes 0-15: col r, K[0..7]+K[16..23]; lanes 16-31: col r, K[8..15]+K[24..31]
static __device__ __forceinline__ v16bf load_b_frag(const bf16_t* base, int ld, int col, int h) {
  const bf16_t* p = base + (size_t)col * ld + h * 8;
  return frag_from(p, p + 16);
}

// CDNA5 async global->LDS staging (ASYNCcnt-tracked), GVS addressing.
static __device__ __forceinline__ void async_ld_b128(uint32_t lds_addr, uint64_t gbase, uint32_t goff) {
  asm volatile("global_load_async_to_lds_b128 %0, %1, %2"
               :: "v"(lds_addr), "v"(goff), "s"(gbase)
               : "memory");
}
static __device__ __forceinline__ void wait_async0() {
  asm volatile("s_wait_asynccnt 0" ::: "memory");
}
static __device__ __forceinline__ void wait_ds0() {
  asm volatile("s_wait_dscnt 0" ::: "memory");
}

// DPP16 ROW_XMASK butterfly (stays within 16-lane rows): pure-VALU reductions.
template <int M>
static __device__ __forceinline__ float row_xor(float v) {
  int t = __builtin_amdgcn_update_dpp(0, __float_as_int(v), 0x160 + M, 0xf, 0xf, true);
  return __int_as_float(t);
}
static __device__ __forceinline__ float row_max16(float v) {
  v = fmaxf(v, row_xor<1>(v));
  v = fmaxf(v, row_xor<2>(v));
  v = fmaxf(v, row_xor<4>(v));
  v = fmaxf(v, row_xor<8>(v));
  return v;
}
static __device__ __forceinline__ float row_sum16(float v) {
  v += row_xor<1>(v);
  v += row_xor<2>(v);
  v += row_xor<4>(v);
  v += row_xor<8>(v);
  return v;
}

// ---------------------------------------------------------------------------
// Kernel 1: fused 1x1 convs + PReLU, pack Q/K pixel-major bf16, V channel-major
// ---------------------------------------------------------------------------
__global__ __launch_bounds__(128) void nla_proj(
    const float* __restrict__ x,
    const float* __restrict__ w1, const float* __restrict__ b1, const float* __restrict__ a1,
    const float* __restrict__ w2, const float* __restrict__ b2, const float* __restrict__ a2,
    const float* __restrict__ w3, const float* __restrict__ b3, const float* __restrict__ a3,
    bf16_t* __restrict__ Q, bf16_t* __restrict__ K, bf16_t* __restrict__ V) {
  __shared__ float xs[128 * 68];   // [pixel][channel], padded stride to spread banks

  const int tid = threadIdx.x;
  const int n = blockIdx.y;
  const int p = blockIdx.x * 128 + tid;
  const float* xn = x + ((size_t)n * CV) * HW;

#pragma unroll 8
  for (int c = 0; c < 64; ++c) xs[tid * 68 + c] = xn[(size_t)c * HW + p];
  __syncthreads();

  const float al1 = a1[0], al2 = a2[0], al3 = a3[0];
  const float* xr = &xs[tid * 68];

  bf16_t* Qp = Q + ((size_t)n * HW + p) * CQK;
  bf16_t* Kp = K + ((size_t)n * HW + p) * CQK;
  bf16_t* Vp = V + ((size_t)n * CV) * HW + p;

#pragma unroll 4
  for (int o = 0; o < 32; ++o) {
    float acc = b1[o];
#pragma unroll
    for (int c4 = 0; c4 < 16; ++c4) {
      float4 xv = *(const float4*)&xr[c4 * 4];
      acc += w1[o * 64 + c4 * 4 + 0] * xv.x + w1[o * 64 + c4 * 4 + 1] * xv.y +
             w1[o * 64 + c4 * 4 + 2] * xv.z + w1[o * 64 + c4 * 4 + 3] * xv.w;
    }
    Qp[o] = (bf16_t)(acc >= 0.0f ? acc : al1 * acc);
  }
#pragma unroll 4
  for (int o = 0; o < 32; ++o) {
    float acc = b2[o];
#pragma unroll
    for (int c4 = 0; c4 < 16; ++c4) {
      float4 xv = *(const float4*)&xr[c4 * 4];
      acc += w2[o * 64 + c4 * 4 + 0] * xv.x + w2[o * 64 + c4 * 4 + 1] * xv.y +
             w2[o * 64 + c4 * 4 + 2] * xv.z + w2[o * 64 + c4 * 4 + 3] * xv.w;
    }
    Kp[o] = (bf16_t)(acc >= 0.0f ? acc : al2 * acc);
  }
#pragma unroll 4
  for (int o = 0; o < 64; ++o) {
    float acc = b3[o];
#pragma unroll
    for (int c4 = 0; c4 < 16; ++c4) {
      float4 xv = *(const float4*)&xr[c4 * 4];
      acc += w3[o * 64 + c4 * 4 + 0] * xv.x + w3[o * 64 + c4 * 4 + 1] * xv.y +
             w3[o * 64 + c4 * 4 + 2] * xv.z + w3[o * 64 + c4 * 4 + 3] * xv.w;
    }
    Vp[(size_t)o * HW] = (bf16_t)(acc >= 0.0f ? acc : al3 * acc);
  }
}

// ---------------------------------------------------------------------------
// Kernel 2: flash attention. WG = 128 queries (8 waves x 16 rows), one batch.
// Double-buffered async K/V staging overlapped with WMMA compute.
// ---------------------------------------------------------------------------
__global__ __launch_bounds__(256) void nla_attn(
    const bf16_t* __restrict__ Q, const bf16_t* __restrict__ K, const bf16_t* __restrict__ V,
    const float* __restrict__ x, float* __restrict__ out) {
  __shared__ bf16_t Ksh[2][BK * CQK];       // [key][ch]   2 x 4 KB
  __shared__ bf16_t Vsh[2][CV * BK];        // [ch][key]   2 x 8 KB
  __shared__ bf16_t Psh[NWAVE * 16 * BK];   // per-wave P  16 KB
  __shared__ float  Tsh[NWAVE][16 * 17];    // epilogue transpose, padded

  const int tid = threadIdx.x;
  const int lane = tid & 31;
  const int wave = tid >> 5;
  const int r = lane & 15;
  const int h = lane >> 4;
  const int n = blockIdx.y;
  const int qbase = blockIdx.x * 128 + wave * 16;

  const bf16_t* Qn = Q + (size_t)n * HW * CQK;
  const bf16_t* Kn = K + (size_t)n * HW * CQK;
  const bf16_t* Vn = V + (size_t)n * CV * HW;

  const v16bf qf = load_a_frag(Qn + (size_t)qbase * CQK, CQK, r, h);

  v8f O[4];
#pragma unroll
  for (int t = 0; t < 4; ++t) O[t] = vzero8();
  v8f m, l;
#pragma unroll
  for (int c = 0; c < 8; ++c) { m[c] = -1e30f; l[c] = 0.0f; }

  const uint32_t ldsK[2] = { (uint32_t)(uintptr_t)&Ksh[0][0], (uint32_t)(uintptr_t)&Ksh[1][0] };
  const uint32_t ldsV[2] = { (uint32_t)(uintptr_t)&Vsh[0][0], (uint32_t)(uintptr_t)&Vsh[1][0] };
  const uint64_t gK = (uint64_t)(uintptr_t)Kn;
  const uint64_t gV = (uint64_t)(uintptr_t)Vn;
  bf16_t* pw = Psh + wave * 16 * BK;

  const int chA = tid >> 3, chB = (tid + 256) >> 3, part = tid & 7;

  // 3 async b128 per thread stage one 64-key K/V block (4 KB + 8 KB).
  auto stage = [&](int buf, int kb) {
    async_ld_b128(ldsK[buf] + tid * 16, gK, (uint32_t)((kb * CQK + tid * 8) * 2));
    async_ld_b128(ldsV[buf] + (uint32_t)((chA * BK + part * 8) * 2), gV,
                  (uint32_t)((chA * HW + kb + part * 8) * 2));
    async_ld_b128(ldsV[buf] + (uint32_t)((chB * BK + part * 8) * 2), gV,
                  (uint32_t)((chB * HW + kb + part * 8) * 2));
  };

  stage(0, 0);  // prologue

  for (int i = 0; i < NBLK; ++i) {
    const int cur = i & 1;
    wait_async0();        // this wave's block-i staging done
    __syncthreads();      // everyone's staging done; prev buffer free for reuse
    if (i + 1 < NBLK) stage(cur ^ 1, (i + 1) * BK);  // overlap next copy with compute

    const bf16_t* Kc = &Ksh[cur][0];
    const bf16_t* Vc = &Vsh[cur][0];

    // --- S = Q K^T : 16 x 64 scores as four 16x16 f32 tiles ---
    v8f s[4];
#pragma unroll
    for (int t = 0; t < 4; ++t) {
      v16bf kf = load_b_frag(Kc, CQK, t * 16 + r, h);
      s[t] = __builtin_amdgcn_wmma_f32_16x16x32_bf16(false, qf, false, kf,
                                                     (short)0, vzero8(), false, false);
    }

    // --- online softmax: row stats via DPP ROW_XMASK butterflies (VALU only) ---
    v8f mb = s[0];
#pragma unroll
    for (int t = 1; t < 4; ++t)
#pragma unroll
      for (int c = 0; c < 8; ++c) mb[c] = fmaxf(mb[c], s[t][c]);
#pragma unroll
    for (int c = 0; c < 8; ++c) mb[c] = row_max16(mb[c]);

    v8f mn, sc, rs;
#pragma unroll
    for (int c = 0; c < 8; ++c) {
      mn[c] = fmaxf(m[c], mb[c]);
      sc[c] = __expf(m[c] - mn[c]);
      rs[c] = 0.0f;
    }
#pragma unroll
    for (int t = 0; t < 4; ++t)
#pragma unroll
      for (int c = 0; c < 8; ++c) {
        float p = __expf(s[t][c] - mn[c]);
        s[t][c] = p;
        rs[c] += p;
      }
#pragma unroll
    for (int c = 0; c < 8; ++c) {
      l[c] = l[c] * sc[c] + row_sum16(rs[c]);
      m[c] = mn[c];
    }
#pragma unroll
    for (int t = 0; t < 4; ++t)
#pragma unroll
      for (int c = 0; c < 8; ++c) O[t][c] *= sc[c];

    // --- reshape P (D-layout -> A-layout) through wave-private LDS ---
#pragma unroll
    for (int t = 0; t < 4; ++t)
#pragma unroll
      for (int c = 0; c < 8; ++c)
        pw[(c + 8 * h) * BK + t * 16 + r] = (bf16_t)s[t][c];
    wait_ds0();

    v16bf pa0 = load_a_frag(pw, BK, r, h);       // keys kb+0..31
    v16bf pa1 = load_a_frag(pw + 32, BK, r, h);  // keys kb+32..63

    // --- O += P V : V channel-major is exactly B^T row-major ---
#pragma unroll
    for (int tc = 0; tc < 4; ++tc) {
      v16bf v0 = load_b_frag(Vc, BK, tc * 16 + r, h);
      v16bf v1 = load_b_frag(Vc + 32, BK, tc * 16 + r, h);
      O[tc] = __builtin_amdgcn_wmma_f32_16x16x32_bf16(false, pa0, false, v0,
                                                      (short)0, O[tc], false, false);
      O[tc] = __builtin_amdgcn_wmma_f32_16x16x32_bf16(false, pa1, false, v1,
                                                      (short)0, O[tc], false, false);
    }
    __syncthreads();  // all waves done reading cur before it is restaged
  }

  // --- epilogue: normalize, LDS-transpose each 16x16 tile so stores map
  //     lanes -> consecutive pixels (coalesced), add residual, store fp32 ---
  v8f linv;
#pragma unroll
  for (int c = 0; c < 8; ++c) linv[c] = 1.0f / l[c];

  float* tw = &Tsh[wave][0];
#pragma unroll
  for (int tc = 0; tc < 4; ++tc) {
    // D-layout (lane=ch, comp=row) -> LDS [row][ch]
#pragma unroll
    for (int c = 0; c < 8; ++c)
      tw[(c + 8 * h) * 17 + r] = O[tc][c] * linv[c];
    wait_ds0();
    // store phase: lane -> q = qbase + r (contiguous), comp -> channel
#pragma unroll
    for (int j = 0; j < 8; ++j) {
      const int ch = tc * 16 + j + 8 * h;
      const size_t idx = ((size_t)n * CV + ch) * HW + qbase + r;
      out[idx] = tw[r * 17 + j + 8 * h] + x[idx];
    }
    wait_ds0();  // tile scratch reused next iteration
  }
}

extern "C" void kernel_launch(void* const* d_in, const int* in_sizes, int n_in,
                              void* d_out, int out_size, void* d_ws, size_t ws_size,
                              hipStream_t stream) {
  (void)in_sizes; (void)n_in; (void)out_size; (void)ws_size;
  const float* x  = (const float*)d_in[0];
  const float* w1 = (const float*)d_in[1];
  const float* b1 = (const float*)d_in[2];
  const float* a1 = (const float*)d_in[3];
  const float* w2 = (const float*)d_in[4];
  const float* b2 = (const float*)d_in[5];
  const float* a2 = (const float*)d_in[6];
  const float* w3 = (const float*)d_in[7];
  const float* b3 = (const float*)d_in[8];
  const float* a3 = (const float*)d_in[9];
  float* out = (float*)d_out;

  bf16_t* Q = (bf16_t*)d_ws;                 // 8*4096*32 bf16 = 2 MB
  bf16_t* K = Q + (size_t)8 * HW * CQK;      // 2 MB
  bf16_t* V = K + (size_t)8 * HW * CQK;      // 4 MB

  nla_proj<<<dim3(32, 8), 128, 0, stream>>>(x, w1, b1, a1, w2, b2, a2, w3, b3, a3, Q, K, V);
  nla_attn<<<dim3(32, 8), 256, 0, stream>>>(Q, K, V, x, out);
}
